// K_attention_ex_12086037971051
// MI455X (gfx1250) — matrix operand order repaired and checked by reading, not compiled, not run on hardware
//
#include <hip/hip_runtime.h>
#include <stdint.h>

// ---------------------------------------------------------------------------
// Gaussian kernel attention: out = x + (exp(-sqdist*r_sigma+margin) ∘ offdiag) @ x
// B=16, T=2048, C=64, f32 in/out.  Fused flash-style: never materialize K.
// Matmuls via v_wmma_f32_16x16x32_bf16 (wave32, CDNA5).
// Double-buffered LDS key tiles (1 barrier/step), preloaded WMMA operands,
// diagonal mask hoisted behind a once-per-64-steps branch.
// d_ws layout: [0, 4MB) x in bf16; [4MB, 4MB+128KB) row sums of squares (f32).
// ---------------------------------------------------------------------------

#define B_DIM 16
#define T_DIM 2048
#define C_DIM 64
#define NSTEP (T_DIM / 32)

typedef __attribute__((ext_vector_type(16))) __bf16 v16bf;
typedef __attribute__((ext_vector_type(8)))  float  v8f;

union VB  { v16bf v; uint4 q[2]; uint2 d[4]; };
union US8 { uint4 q; unsigned short s[8]; };

static __device__ __forceinline__ unsigned short f32_to_bf16(float f) {
  union { float f; unsigned u; } v; v.f = f;
  unsigned r = (v.u + 0x7FFFu + ((v.u >> 16) & 1u)) >> 16;   // RNE
  return (unsigned short)r;
}

static __device__ __forceinline__ v8f wmma_bf16(const VB& a, const VB& b, v8f c) {
  return __builtin_amdgcn_wmma_f32_16x16x32_bf16(false, a.v, false, b.v,
                                                 (short)0, c, false, false);
}

// One row (64 channels) per wave: bf16 copy + sum of squares.
__global__ __launch_bounds__(128) void K_prepass(const float* __restrict__ x,
                                                 unsigned* __restrict__ xbf,
                                                 float* __restrict__ sq) {
  const int row  = blockIdx.x * 4 + (threadIdx.x >> 5);
  const int lane = threadIdx.x & 31;
  const float2 v = *(const float2*)(x + (size_t)row * C_DIM + lane * 2);
  float ss = v.x * v.x + v.y * v.y;
#pragma unroll
  for (int m = 16; m >= 1; m >>= 1) ss += __shfl_xor(ss, m, 32);
  xbf[(size_t)row * 32 + lane] =
      (unsigned)f32_to_bf16(v.x) | ((unsigned)f32_to_bf16(v.y) << 16);
  if (lane == 0) sq[row] = ss;
}

struct TileRegs { US8 d0, d1; float sqv; };

__global__ __launch_bounds__(128) void K_gauss_attn(
    const float* __restrict__ x, const unsigned short* __restrict__ xbf,
    const float* __restrict__ sqg, const float* __restrict__ rsig,
    const float* __restrict__ marg, float* __restrict__ out) {
  // Double-buffered key tiles.
  // Row-major (32 keys x 64ch), row padded to 72 bf16 (conflict-free b128 reads)
  __shared__ __align__(16) unsigned short Kl[2][32 * 72];
  // Transposed (64ch x 32 keys), row padded to 36 bf16 (conflict-free b64 reads)
  __shared__ __align__(16) unsigned short Vl[2][64 * 36];
  __shared__ float sqj[2][32];
  // Per-wave W (16q x 32j), row padded to 48 bf16 (aligned b128 A-tile reads)
  __shared__ __align__(16) unsigned short Wl[4][16 * 48];

  const int tid  = threadIdx.x;
  const int wave = tid >> 5;
  const int lane = tid & 31;
  const int h    = lane >> 4;   // half-wave select (WMMA layouts)
  const int ln   = lane & 15;

  const int b     = blockIdx.x >> 5;                     // batch
  const int qbase = (blockIdx.x & 31) * 64 + wave * 16;  // query row in batch
  const int diagBase = qbase & ~31;                      // only step with diagonal

  const float LOG2E = 1.4426950408889634f;
  const float c1 = -rsig[0] * LOG2E;            // exp(z) = exp2(z*log2e)
  const float c2 =  marg[0] * LOG2E;

  const size_t batchOff = (size_t)b * T_DIM * C_DIM;
  const unsigned short* xb = xbf + batchOff;
  const float* sqb = sqg + b * T_DIM;

  // Resident Q A-tiles (16x32 bf16 each, channels [0,32) and [32,64)).
  const size_t rowQ = batchOff + (size_t)(qbase + ln) * C_DIM;
  const int koff = h * 16;
  VB a0, a1;
  a0.q[0] = *(const uint4*)(xbf + rowQ + koff);
  a0.q[1] = *(const uint4*)(xbf + rowQ + 8 + koff);
  a1.q[0] = *(const uint4*)(xbf + rowQ + 32 + koff);
  a1.q[1] = *(const uint4*)(xbf + rowQ + 40 + koff);

  float sqi[8]; int qg[8];
#pragma unroll
  for (int r = 0; r < 8; ++r) {                 // C/D layout: M = r + 8*h
    qg[r]  = qbase + r + 8 * h;
    sqi[r] = sqb[qg[r]];
  }

  v8f o[4];                                     // O tile 16x64, f32 accum
#pragma unroll
  for (int i = 0; i < 4; ++i) o[i] = (v8f){};

  const int jrow = tid >> 2;                    // staging assignment
  const int c0   = (tid & 3) * 16;

  auto load_tile = [&](int jbase, TileRegs& t) {
    const unsigned short* gp = xb + (size_t)(jbase + jrow) * C_DIM + c0;
    t.d0.q = *(const uint4*)gp;
    t.d1.q = *(const uint4*)(gp + 8);
    t.sqv  = sqb[jbase + (tid & 31)];
  };
  auto store_tile = [&](int p, const TileRegs& t) {
    *(uint4*)&Kl[p][jrow * 72 + c0]     = t.d0.q;
    *(uint4*)&Kl[p][jrow * 72 + c0 + 8] = t.d1.q;
#pragma unroll
    for (int i = 0; i < 8; ++i) {
      Vl[p][(c0 + i)     * 36 + jrow] = t.d0.s[i];
      Vl[p][(c0 + 8 + i) * 36 + jrow] = t.d1.s[i];
    }
    if (tid < 32) sqj[p][tid] = t.sqv;
  };

  { TileRegs t0; load_tile(0, t0); store_tile(0, t0); }   // prologue: tile 0

  for (int s = 0; s < NSTEP; ++s) {
    __syncthreads();            // tile s visible; buffer p^1 free to overwrite
    const int p = s & 1;
    const int jbase = s * 32;

    TileRegs nxt;
    if (s < NSTEP - 1) load_tile(jbase + 32, nxt);  // overlap with compute

    const unsigned short* Kp = Kl[p];
    const unsigned short* Vp = Vl[p];

    // ---- Gram: preload all four B tiles, then WMMA chain ----
    VB bb[4];   // [jsub*2 + chalf]
#pragma unroll
    for (int jsub = 0; jsub < 2; ++jsub) {
      const int jr = jsub * 16 + ln;            // B layout: lane = N = key
      bb[2 * jsub + 0].q[0] = *(const uint4*)&Kp[jr * 72 + h * 16];
      bb[2 * jsub + 0].q[1] = *(const uint4*)&Kp[jr * 72 + h * 16 + 8];
      bb[2 * jsub + 1].q[0] = *(const uint4*)&Kp[jr * 72 + 32 + h * 16];
      bb[2 * jsub + 1].q[1] = *(const uint4*)&Kp[jr * 72 + 32 + h * 16 + 8];
    }
    v8f s0 = (v8f){}, s1 = (v8f){};
    s0 = wmma_bf16(a0, bb[0], s0);
    s0 = wmma_bf16(a1, bb[1], s0);
    s1 = wmma_bf16(a0, bb[2], s1);
    s1 = wmma_bf16(a1, bb[3], s1);

    // ---- Gaussian weights ----
    const float sj0 = sqj[p][ln];
    const float sj1 = sqj[p][16 + ln];
    float w[16];
#pragma unroll
    for (int r = 0; r < 8; ++r) {
      float d0 = fmaxf(sqi[r] + sj0 - 2.0f * s0[r], 0.0f);
      float d1 = fmaxf(sqi[r] + sj1 - 2.0f * s1[r], 0.0f);
      w[r]     = __builtin_amdgcn_exp2f(fmaf(d0, c1, c2));
      w[8 + r] = __builtin_amdgcn_exp2f(fmaf(d1, c1, c2));
    }
    if (jbase == diagBase) {                    // rare path: 1 of 64 steps
      const int jgl0 = jbase + ln, jgl1 = jbase + 16 + ln;
#pragma unroll
      for (int r = 0; r < 8; ++r) {
        if (qg[r] == jgl0) w[r] = 0.0f;
        if (qg[r] == jgl1) w[8 + r] = 0.0f;
      }
    }
#pragma unroll
    for (int r = 0; r < 8; ++r) {
      Wl[wave][(r + 8 * h) * 48 + ln]      = f32_to_bf16(w[r]);
      Wl[wave][(r + 8 * h) * 48 + 16 + ln] = f32_to_bf16(w[8 + r]);
    }
    // Per-wave LDS transpose round-trip: lanes are lockstep, just drain DS.
    asm volatile("s_wait_dscnt 0" ::: "memory");

    // ---- AV: W as A-matrix (16x32, M=q, K=j); preload all B tiles ----
    VB wa;
    wa.q[0] = *(const uint4*)&Wl[wave][ln * 48 + koff];
    wa.q[1] = *(const uint4*)&Wl[wave][ln * 48 + 8 + koff];
    VB vbs[4];
#pragma unroll
    for (int cb = 0; cb < 4; ++cb) {            // B: lane = N = channel, K = key
      const int c = cb * 16 + ln;
#pragma unroll
      for (int k = 0; k < 4; ++k)
        vbs[cb].d[k] = *(const uint2*)&Vp[c * 36 + h * 16 + k * 4];
    }
#pragma unroll
    for (int cb = 0; cb < 4; ++cb)
      o[cb] = wmma_bf16(wa, vbs[cb], o[cb]);

    if (s < NSTEP - 1) store_tile(p ^ 1, nxt);  // drain prefetch into other buffer
  }

  // Epilogue: residual add + store.
#pragma unroll
  for (int cb = 0; cb < 4; ++cb) {
#pragma unroll
    for (int r = 0; r < 8; ++r) {
      const size_t idx =
          batchOff + (size_t)(qbase + r + 8 * h) * C_DIM + cb * 16 + ln;
      out[idx] = x[idx] + o[cb][r];
    }
  }
}

extern "C" void kernel_launch(void* const* d_in, const int* in_sizes, int n_in,
                              void* d_out, int out_size, void* d_ws, size_t ws_size,
                              hipStream_t stream) {
  (void)in_sizes; (void)n_in; (void)out_size; (void)ws_size;
  const float* x  = (const float*)d_in[0];
  const float* rs = (const float*)d_in[1];
  const float* mg = (const float*)d_in[2];
  float* out = (float*)d_out;

  unsigned short* xbf = (unsigned short*)d_ws;                       // 4 MB
  float* sq = (float*)((char*)d_ws + (size_t)B_DIM * T_DIM * C_DIM * 2);

  K_prepass<<<(B_DIM * T_DIM) / 4, 128, 0, stream>>>(x, (unsigned*)xbf, sq);
  K_gauss_attn<<<B_DIM * (T_DIM / 64), 128, 0, stream>>>(x, xbf, sq, rs, mg, out);
}
